// HBondSymmetric_549755813996
// MI455X (gfx1250) — matrix operand order repaired and checked by reading, not compiled, not run on hardware
//
#include <hip/hip_runtime.h>

typedef _Float16 half_t;
typedef __attribute__((ext_vector_type(2)))  half_t h2;
typedef __attribute__((ext_vector_type(16))) half_t v16h;
typedef __attribute__((ext_vector_type(8)))  float  v8f;
typedef __attribute__((ext_vector_type(4)))  int    v4i;

// pointer-to-addrspace-qualified-int4 types for the async LDS DMA builtin
typedef __attribute__((address_space(1))) v4i* gas_p;   // global
typedef __attribute__((address_space(3))) v4i* las_p;   // LDS

union AU { v16h v; h2 h[8]; };

#define PI_F 3.14159265358979323846f

// Exact K = 48*48 = 2304 -> 72 chunks of 32 per term.
#define NCHUNK 72
#define TERM_HALVES (NCHUNK * 512)                   // 36864 halves = 73728 B
#define SENSE_BYTES (8 * 32 * 16 * 4)                // 16384
#define B_BYTES     (TERM_HALVES * 2 * 2)            // both terms: 147456
#define SMEM_BYTES  (SENSE_BYTES + B_BYTES)          // 163840 (160 KB of 320 KB WGP LDS)

// ---------------------------------------------------------------------------
// Prep: build B (f16) in WMMA B-operand lane layout, exact K ordering k=f*48+g.
//   wB[t][c][hb][s][i],  K = c*32 + (i&8)*2 + hb*8 + (i&7),  f = K/48, g = K%48
//   B[k,s] = Wsym[s,f,g] = w[s,0,f,g] + w[s,0,g,f]
// ---------------------------------------------------------------------------
__global__ void prep_wB(const float* __restrict__ w0,
                        const float* __restrict__ w1,
                        half_t* __restrict__ wB) {
  int idx = blockIdx.x * 256 + threadIdx.x;
  if (idx >= 2 * NCHUNK * 512) return;
  int i  = idx & 15;
  int s  = (idx >> 4) & 15;
  int hb = (idx >> 8) & 1;
  int r  = idx >> 9;
  int c  = r % NCHUNK;
  int t  = r / NCHUNK;
  int K  = c * 32 + (i & 8) * 2 + hb * 8 + (i & 7);
  int f  = K / 48;
  int g  = K % 48;
  const float* w = t ? w1 : w0;
  float v = w[s * 2304 + f * 48 + g] + w[s * 2304 + g * 48 + f];
  wB[idx] = (half_t)v;
}

// ---------------------------------------------------------------------------
// Feature-row loaders (f32 -> packed f16 pairs).
// ---------------------------------------------------------------------------
__device__ __forceinline__ void load_xf(const float* __restrict__ p, h2* xf) {
#pragma unroll
  for (int i = 0; i < 24; ++i) {
    float2 v = ((const float2*)p)[i];
    xf[i] = h2{(half_t)v.x, (half_t)v.y};
  }
}

// xg pre-shifted by lane half: xg[i] = g-pair (i + 4*hb), i = 0..19.
__device__ __forceinline__ void load_xg(const float* __restrict__ p, int hb, h2* xg) {
  const float2* q = (const float2*)p + 4 * hb;
#pragma unroll
  for (int i = 0; i < 20; ++i) {
    float2 v = q[i];
    xg[i] = h2{(half_t)v.x, (half_t)v.y};
  }
}

// Per-(chunk%3, j) A-build tables (ISA A layout, hb pre-shift folded into xg).
__constant__ const int kIdxTab[3][8] = {
  { 0, 1, 2, 3,  8, 9,10,11},   // cg0: f0 g0..15 / g16..31
  {16,17,18,19,  0, 1, 2, 3},   // cg1: f0 g32..47 | f1 g0..15
  { 8, 9,10,11, 16,17,18,19}};  // cg2: f1 g16..31 / g32..47
__constant__ const int kFTab[3][8] = {
  {0,0,0,0, 0,0,0,0},
  {0,0,0,0, 1,1,1,1},
  {1,1,1,1, 1,1,1,1}};

__device__ __forceinline__ void wait_async0() {
#if __has_builtin(__builtin_amdgcn_s_wait_asynccnt)
  __builtin_amdgcn_s_wait_asynccnt(0);
#else
  asm volatile("s_wait_asynccnt 0x0" ::: "memory");
#endif
}

// ---------------------------------------------------------------------------
// Main kernel: wave = 32 pairs (two 16-row M-tiles), N = 16 sensitivities,
// K = 2304 outer-product dim via v_wmma_f32_16x16x32_f16, B resident in LDS.
// ---------------------------------------------------------------------------
__global__ __launch_bounds__(256) void pair_kernel(
    const float* __restrict__ feat0, const float* __restrict__ feat1,
    const half_t* __restrict__ wB,
    const float* __restrict__ mu, const float* __restrict__ sigma,
    const float* __restrict__ pair_dist,
    const int* __restrict__ pair_first, const int* __restrict__ pair_second,
    float* __restrict__ out, int P) {
  extern __shared__ char smem_raw[];
  float*  sSense = (float*)smem_raw;                    // [8][32][16]
  half_t* sB     = (half_t*)(smem_raw + SENSE_BYTES);   // both terms' B

  const int tid  = threadIdx.x;
  const int lane = tid & 31;
  const int wave = tid >> 5;
  const int hb   = (lane >> 4) & 1;
  const int sL   = lane & 15;
  const int base = (blockIdx.x * 8 + wave) * 32;

  // --- async DMA both terms' B: global -> LDS (tracked by ASYNCcnt) --------
#if __has_builtin(__builtin_amdgcn_global_load_async_to_lds_b128)
  {
    char* gsrc = (char*)wB;           // builtin takes non-const v4i AS(1)*
    char* ldst = (char*)sB;
#pragma unroll
    for (int k = 0; k < B_BYTES / 16 / 256; ++k) {      // 36 iters
      int off = (k * 256 + tid) * 16;
      __builtin_amdgcn_global_load_async_to_lds_b128(
          (gas_p)(gsrc + off), (las_p)(ldst + off), 0, 0);
    }
  }
#else
  {
    const uint4* src = (const uint4*)wB;
    uint4* dst = (uint4*)sB;
#pragma unroll
    for (int k = 0; k < B_BYTES / 16 / 256; ++k)
      dst[k * 256 + tid] = src[k * 256 + tid];
  }
#endif

  // --- sensitivities: one pair per lane, 16 s each -------------------------
  {
    int q = base + lane; if (q > P - 1) q = P - 1;
    float d   = pair_dist[q];
    float inv = 1.f / d;
    float cut = 0.5f * (__cosf(PI_F * d * (1.f / 6.f)) + 1.f);
    cut = (d < 6.f) ? cut : 0.f;
#pragma unroll
    for (int s = 0; s < 16; ++s) {
      float nd = (inv - mu[s]) / sigma[s];
      sSense[(wave * 32 + lane) * 16 + s] = __expf(-0.5f * nd * nd) * cut;
    }
  }

  int m0 = base + sL;      if (m0 > P - 1) m0 = P - 1;
  int m1 = base + 16 + sL; if (m1 > P - 1) m1 = P - 1;
  const int a0 = pair_first[m0],  b0 = pair_second[m0];
  const int a1 = pair_first[m1],  b1 = pair_second[m1];

#if __has_builtin(__builtin_amdgcn_global_load_async_to_lds_b128)
  wait_async0();
#endif
  __syncthreads();   // B + sense visible to all waves

  float pt0 = 0.f, pt1 = 0.f;

#pragma unroll
  for (int term = 0; term < 2; ++term) {
    const half_t* pBl = sB + term * TERM_HALVES + hb * 256 + sL * 16;

    const float* feat = term ? feat1 : feat0;
    h2 xf0[24], xf1[24], xg0[20], xg1[20];
    load_xf(feat + (size_t)a0 * 48, xf0);
    load_xf(feat + (size_t)a1 * 48, xf1);
    load_xg(feat + (size_t)b0 * 48, hb, xg0);
    load_xg(feat + (size_t)b1 * 48, hb, xg1);

    v8f acc0 = {}, acc1 = {};
    AU A0[3], A1[3];      // 3-deep A staging: keeps WAR writers outside the
                          // 4-coexec WMMA hazard window
    v16h bvb[2];          // 2-deep B prefetch from LDS

    auto BUILD = [&](int c, int buf) {
      const int grp = c / 3, cg = c % 3;
      const h2 xe0 = h2{xf0[grp][0], xf0[grp][0]};
      const h2 xo0 = h2{xf0[grp][1], xf0[grp][1]};
      const h2 xe1 = h2{xf1[grp][0], xf1[grp][0]};
      const h2 xo1 = h2{xf1[grp][1], xf1[grp][1]};
#pragma unroll
      for (int j = 0; j < 8; ++j) {
        const int gi = kIdxTab[cg][j];
        A0[buf].h[j] = (kFTab[cg][j] ? xo0 : xe0) * xg0[gi];
        A1[buf].h[j] = (kFTab[cg][j] ? xo1 : xe1) * xg1[gi];
      }
    };

    BUILD(0, 0);
    bvb[0] = *(const v16h*)(pBl + (0 << 9));
#pragma unroll
    for (int c = 0; c < NCHUNK; ++c) {
      const int cb = c % 3, nb = (c + 1) % 3;
      const int bb = c & 1, bn = bb ^ 1;
      if (c + 1 < NCHUNK) {
        BUILD(c + 1, nb);                              // next A (VALU)
        bvb[bn] = *(const v16h*)(pBl + ((c + 1) << 9)); // next B (2x ds_load_b128)
      }
      acc0 = __builtin_amdgcn_wmma_f32_16x16x32_f16(false, A0[cb].v, false, bvb[bb],
                                                    (short)0, acc0, false, false);
      acc1 = __builtin_amdgcn_wmma_f32_16x16x32_f16(false, A1[cb].v, false, bvb[bb],
                                                    (short)0, acc1, false, false);
    }

    // --- epilogue: out[b] = sum_s sense[b,s] * C[b,s] ------------------------
    float keep = 0.f;
#pragma unroll
    for (int r = 0; r < 8; ++r) {
      int M = r + 8 * hb;
      float v0 = acc0[r] * sSense[(wave * 32 + M) * 16 + sL];
      v0 += __shfl_xor(v0, 1); v0 += __shfl_xor(v0, 2);
      v0 += __shfl_xor(v0, 4); v0 += __shfl_xor(v0, 8);
      if (sL == r) keep = v0;
      float v1 = acc1[r] * sSense[(wave * 32 + 16 + M) * 16 + sL];
      v1 += __shfl_xor(v1, 1); v1 += __shfl_xor(v1, 2);
      v1 += __shfl_xor(v1, 4); v1 += __shfl_xor(v1, 8);
      if (sL == r + 8) keep = v1;
    }
    if (term == 0) pt0 = keep; else pt1 = keep;
  }

  int pair = base + ((sL & 8) ? 16 : 0) + (sL & 7) + 8 * hb;
  if (pair < P) {
    float total = pt0 + pt1;
    float b0sq = pt0 * pt0, b1sq = pt1 * pt1;
    out[pair] = total;
    out[(size_t)P + pair] = b1sq / (b1sq + b0sq);
  }
}

// ---------------------------------------------------------------------------
extern "C" void kernel_launch(void* const* d_in, const int* in_sizes, int n_in,
                              void* d_out, int out_size, void* d_ws, size_t ws_size,
                              hipStream_t stream) {
  (void)n_in; (void)out_size; (void)ws_size;
  const float* feat0 = (const float*)d_in[0];
  const float* feat1 = (const float*)d_in[1];
  const float* w0    = (const float*)d_in[2];
  const float* w1    = (const float*)d_in[3];
  const float* mu    = (const float*)d_in[4];
  const float* sigma = (const float*)d_in[5];
  const float* pdist = (const float*)d_in[6];
  const int*   pfirst  = (const int*)d_in[7];
  const int*   psecond = (const int*)d_in[8];
  float* out = (float*)d_out;
  const int P = in_sizes[6];

  half_t* wB = (half_t*)d_ws;   // 2 * 36864 halves = 147456 bytes

  prep_wB<<<288, 256, 0, stream>>>(w0, w1, wB);

  int blocks = (P + 255) / 256;   // 8 waves/block * 32 pairs/wave
  pair_kernel<<<blocks, 256, SMEM_BYTES, stream>>>(feat0, feat1, wB, mu, sigma,
                                                   pdist, pfirst, psecond, out, P);
}
